// AttentionWithBias_38457137168470
// MI455X (gfx1250) — matrix operand order repaired and checked by reading, not compiled, not run on hardware
//
#include <hip/hip_runtime.h>
#include <hip/hip_bf16.h>
#include <math.h>

typedef __attribute__((ext_vector_type(16))) __bf16        v16bf;
typedef __attribute__((ext_vector_type(8)))  float         v8f;
typedef __attribute__((ext_vector_type(4)))  unsigned int  u32x4;

#define D_IN   256
#define D_BIAS 128
#define NH     8
#define DH     32
#define BATCH  2
#define SEQ    512
#define NROW   (BATCH * SEQ)                 // 1024 rows of x
#define QK_SCALE 0.17677669529663687f        // 1/sqrt(32)
#define BIAS_ROWS 8                          // rows per wave in k_bias

// ---------------- CDNA5 async global->LDS path (guarded) ----------------
#if defined(__has_builtin)
#if __has_builtin(__builtin_amdgcn_global_load_async_to_lds_b128)
#define HAVE_ASYNC_LDS 1
#endif
#endif

#ifdef HAVE_ASYNC_LDS
// Exact parameter types per hipcc diagnostic: pointee is GCC-vector int4;
// global source is addrspace(1) ("__device__"), LDS dest is addrspace(3).
typedef int v4i_vs __attribute__((vector_size(16)));
typedef __attribute__((address_space(1))) v4i_vs* as1_v4i;
typedef __attribute__((address_space(3))) v4i_vs* as3_v4i;
static __device__ __forceinline__ void copy16_async(const float* g, float* l) {
  __builtin_amdgcn_global_load_async_to_lds_b128((as1_v4i)g, (as3_v4i)l, 0, 0);
}
#if __has_builtin(__builtin_amdgcn_s_wait_asynccnt)
#define WAIT_ASYNC(n) __builtin_amdgcn_s_wait_asynccnt(n)
#else
#define WAIT_ASYNC(n) asm volatile("s_wait_asynccnt %0" :: "i"(n) : "memory")
#endif
#else
static __device__ __forceinline__ void copy16_async(const float* g, float* l) {
  *(u32x4*)l = *(const u32x4*)g;            // sync fallback through VGPRs
}
#define WAIT_ASYNC(n) ((void)0)
#endif

// ---------------------------------------------------------------- helpers
static __device__ __forceinline__ float nan_fix(float v) {
  if (__builtin_isnan(v)) return 0.0f;
  if (__builtin_isinf(v)) return v > 0.0f ? 3.4028234663852886e38f
                                          : -3.4028234663852886e38f;
  return v;
}

union bfpack { v16bf v; u32x4 q[2]; };

// A fragment (16x32 bf16), X row-major [*, ld]; tile origin (m0, k0).
// Lane l: row m0+(l&15); lanes 0-15: K = k0+{0..7,16..23}, lanes 16-31: +8.
// (ISA 7.12.2, 16-bit A 16x32). Two b128 loads per lane.
static __device__ __forceinline__ v16bf load_A16x32(const __bf16* __restrict__ X,
                                                    int ld, int m0, int k0) {
  int lane = threadIdx.x & 31;
  int m  = m0 + (lane & 15);
  int kb = k0 + ((lane >> 4) << 3);
  const __bf16* p = X + (size_t)m * ld + kb;
  bfpack u;
  u.q[0] = *(const u32x4*)(p);
  u.q[1] = *(const u32x4*)(p + 16);
  return u.v;
}

// B fragment (32x16 bf16) where B[k][n] = Xt[n][k], Xt row-major [N, ld]
// (i.e. Xt is the transpose of the mathematical B). Lane l: column n0+(l&15),
// K = k0 + {0..7,16..23}, lanes 16-31: +8 (matches A) -> two b128 loads.
static __device__ __forceinline__ v16bf load_Brow(const __bf16* __restrict__ Xt,
                                                  int ld, int n0, int k0) {
  int lane = threadIdx.x & 31;
  int n  = n0 + (lane & 15);
  int kb = k0 + ((lane >> 4) << 3);
  const __bf16* p = Xt + (size_t)n * ld + kb;
  bfpack u;
  u.q[0] = *(const u32x4*)(p);
  u.q[1] = *(const u32x4*)(p + 16);
  return u.v;
}

static __device__ __forceinline__ v8f wmma_bf16(v16bf a, v16bf b, v8f c) {
  return __builtin_amdgcn_wmma_f32_16x16x32_bf16(false, a, false, b,
                                                 (short)0, c, false, false);
}

// ---------------------------------------------------------------- kernels

// LayerNorm(x) -> f32 + bf16. One 256-thread block per row.
__global__ void k_ln_x(const float* __restrict__ x, const float* __restrict__ g,
                       const float* __restrict__ bvec, float* __restrict__ xn,
                       __bf16* __restrict__ xnb) {
  __shared__ float red[D_IN];
  int row = blockIdx.x, t = threadIdx.x;
  float v = nan_fix(x[(size_t)row * D_IN + t]);
  red[t] = v;
  __syncthreads();
  for (int s = 128; s > 0; s >>= 1) { if (t < s) red[t] += red[t + s]; __syncthreads(); }
  float mean = red[0] * (1.0f / D_IN);
  __syncthreads();
  float d = v - mean;
  red[t] = d * d;
  __syncthreads();
  for (int s = 128; s > 0; s >>= 1) { if (t < s) red[t] += red[t + s]; __syncthreads(); }
  float var = red[0] * (1.0f / D_IN);
  float y = d * rsqrtf(var + 1e-5f) * g[t] + bvec[t];
  xn[(size_t)row * D_IN + t]  = y;
  xnb[(size_t)row * D_IN + t] = (__bf16)y;
}

// Convert the five 256x256 weight matrices to bf16, TRANSPOSED, so that
// WMMA B fragments become contiguous per-lane b128 loads.
__global__ void k_cvt5t(const float* __restrict__ s0, const float* __restrict__ s1,
                        const float* __restrict__ s2, const float* __restrict__ s3,
                        const float* __restrict__ s4,
                        __bf16* __restrict__ d0, __bf16* __restrict__ d1,
                        __bf16* __restrict__ d2, __bf16* __restrict__ d3,
                        __bf16* __restrict__ d4, int n) {
  int i = blockIdx.x * blockDim.x + threadIdx.x;
  if (i >= n) return;
  int k = i >> 8, c = i & 255;          // source row-major [K=256, N=256]
  int it = c * D_IN + k;                // dest [N, K]
  d0[it] = (__bf16)s0[i]; d1[it] = (__bf16)s1[i]; d2[it] = (__bf16)s2[i];
  d3[it] = (__bf16)s3[i]; d4[it] = (__bf16)s4[i];
}

// Q/K/V/G projections: xn[1024,256] @ W[256,256], one wave per 16x16 tile.
// Wt arguments are transposed bf16 weights. V is written TRANSPOSED
// ([B,H,DH,SEQ]) so the PV B fragment in flash attention is contiguous.
__global__ void k_qkvg(const __bf16* __restrict__ xnb, const __bf16* __restrict__ wqt,
                       const __bf16* __restrict__ wkt, const __bf16* __restrict__ wvt,
                       const __bf16* __restrict__ wgt, const float* __restrict__ bg,
                       __bf16* __restrict__ qb, __bf16* __restrict__ kb,
                       __bf16* __restrict__ vt, float* __restrict__ gate) {
  int mt = blockIdx.x, nt = blockIdx.y, which = blockIdx.z;
  const __bf16* Wt = (which == 0) ? wqt : (which == 1) ? wkt
                   : (which == 2) ? wvt : wgt;
  v8f c = {};
#pragma unroll
  for (int k0 = 0; k0 < D_IN; k0 += 32) {
    v16bf a = load_A16x32(xnb, D_IN, mt * 16, k0);
    v16bf b = load_Brow(Wt, D_IN, nt * 16, k0);
    c = wmma_bf16(a, b, c);
  }
  int lane = threadIdx.x & 31;
  int n  = nt * 16 + (lane & 15);
  int mh = (lane >> 4) << 3;
  int h = n >> 5, d = n & 31;
#pragma unroll
  for (int r = 0; r < 8; ++r) {
    int row = mt * 16 + mh + r;
    int bb = row >> 9, ll = row & (SEQ - 1);
    float val = c[r];
    if (which == 0) {
      qb[(((size_t)bb * NH + h) * SEQ + ll) * DH + d] = (__bf16)val;
    } else if (which == 1) {
      kb[(((size_t)bb * NH + h) * SEQ + ll) * DH + d] = (__bf16)(val * QK_SCALE);
    } else if (which == 2) {
      vt[(((size_t)bb * NH + h) * DH + d) * SEQ + ll] = (__bf16)val;   // transposed
    } else {
      gate[(size_t)row * D_IN + n] = 1.0f / (1.0f + __expf(-(val + bg[n])));
    }
  }
}

// Stream bias [B*L*L, 128]: nan_to_num -> LN(128) -> @ Wb[128,8] -> bias_h.
// This kernel IS the roofline (268 MB @ 23.3 TB/s ~ 11.5 us).
// Double-buffered async global->LDS per wave: each wave streams BIAS_ROWS
// consecutive 512B rows via GLOBAL_LOAD_ASYNC_TO_LDS_B128, processing row i
// from LDS while row i+1 is in flight (ASYNCcnt pipelining).
// Output layout [B, H, L, L] so the attention kernel reads it coalesced.
__global__ void k_bias(const float* __restrict__ bias, const float* __restrict__ g,
                       const float* __restrict__ bvec, const float* __restrict__ Wb,
                       float* __restrict__ bh) {
  __shared__ float sWb[D_BIAS * NH];
  __shared__ float sg[D_BIAS], sb[D_BIAS];
  __shared__ __align__(16) float sbuf[8][2][D_BIAS];
  int t = threadIdx.x;
  for (int i = t; i < D_BIAS * NH; i += blockDim.x) sWb[i] = Wb[i];
  for (int i = t; i < D_BIAS; i += blockDim.x) { sg[i] = g[i]; sb[i] = bvec[i]; }
  __syncthreads();

  int wv = t >> 5, lane = t & 31;
  size_t row0 = ((size_t)blockIdx.x * 8 + wv) * BIAS_ROWS;

  // prime buffer 0
  copy16_async(bias + row0 * D_BIAS + lane * 4, &sbuf[wv][0][lane * 4]);

  for (int it = 0; it < BIAS_ROWS; ++it) {
    size_t row = row0 + it;
    if (it + 1 < BIAS_ROWS) {
      copy16_async(bias + (row + 1) * D_BIAS + lane * 4,
                   &sbuf[wv][(it + 1) & 1][lane * 4]);
      WAIT_ASYNC(1);                    // current row complete, prefetch in flight
    } else {
      WAIT_ASYNC(0);
    }
    asm volatile("" ::: "memory");      // keep LDS reads below the wait

    const float* p = &sbuf[wv][it & 1][lane * 4];
    float y0 = nan_fix(p[0]), y1 = nan_fix(p[1]);
    float y2 = nan_fix(p[2]), y3 = nan_fix(p[3]);
    float s  = y0 + y1 + y2 + y3;
    float sq = y0 * y0 + y1 * y1 + y2 * y2 + y3 * y3;
#pragma unroll
    for (int off = 16; off; off >>= 1) {
      s  += __shfl_xor(s,  off, 32);
      sq += __shfl_xor(sq, off, 32);
    }
    float mean = s * (1.0f / D_BIAS);
    float var  = sq * (1.0f / D_BIAS) - mean * mean;
    float inv  = rsqrtf(var + 1e-5f);
    int j = lane * 4;
    y0 = (y0 - mean) * inv * sg[j + 0] + sb[j + 0];
    y1 = (y1 - mean) * inv * sg[j + 1] + sb[j + 1];
    y2 = (y2 - mean) * inv * sg[j + 2] + sb[j + 2];
    y3 = (y3 - mean) * inv * sg[j + 3] + sb[j + 3];
    float acc[NH];
#pragma unroll
    for (int h = 0; h < NH; ++h)
      acc[h] = y0 * sWb[(j + 0) * NH + h] + y1 * sWb[(j + 1) * NH + h] +
               y2 * sWb[(j + 2) * NH + h] + y3 * sWb[(j + 3) * NH + h];
#pragma unroll
    for (int off = 16; off; off >>= 1)
#pragma unroll
      for (int h = 0; h < NH; ++h) acc[h] += __shfl_xor(acc[h], off, 32);
    if (lane == 0) {
      int bb = (int)(row >> 18);
      int q  = (int)(row >> 9) & (SEQ - 1);
      int kc = (int)row & (SEQ - 1);
#pragma unroll
      for (int h = 0; h < NH; ++h)
        bh[(((size_t)bb * NH + h) * SEQ + q) * SEQ + kc] = acc[h];
    }
  }
}

// Flash attention: one wave per (b, h, 16-row q tile). K-loop in steps of 32.
__global__ void k_attn(const __bf16* __restrict__ qb, const __bf16* __restrict__ kbuf,
                       const __bf16* __restrict__ vt, const float* __restrict__ bh,
                       const float* __restrict__ gate, const int* __restrict__ mask,
                       __bf16* __restrict__ go) {
  int qt = blockIdx.x, h = blockIdx.y, b = blockIdx.z;
  int lane = threadIdx.x & 31;
  int half = lane >> 4, nn = lane & 15, mh = half << 3;
  __shared__ __align__(16) __bf16 pl[16 * 32];

  const __bf16* Q  = qb   + ((size_t)b * NH + h) * SEQ * DH;
  const __bf16* K  = kbuf + ((size_t)b * NH + h) * SEQ * DH;
  const __bf16* Vt = vt   + ((size_t)b * NH + h) * DH * SEQ;   // [DH, SEQ]
  const float*  Bh = bh   + ((size_t)b * NH + h) * SEQ * SEQ;  // [q, k]

  v16bf qa = load_A16x32(Q, DH, qt * 16, 0);
  bool mq[8];
#pragma unroll
  for (int r = 0; r < 8; ++r) mq[r] = (mask[b * SEQ + qt * 16 + mh + r] == 0);

  float m[8], lsum[8];
  v8f o0 = {}, o1 = {};
#pragma unroll
  for (int r = 0; r < 8; ++r) { m[r] = -3.0e38f; lsum[r] = 0.0f; }

  for (int kt = 0; kt < SEQ; kt += 32) {
    // scores: Q (16x32) x K^T (32x32) -> two 16x16 tiles. K rows are the
    // mathematical B columns, so load_Brow on K's row-major storage is K^T.
    v16bf kb0 = load_Brow(K, DH, kt, 0);
    v16bf kb1 = load_Brow(K, DH, kt + 16, 0);
    v8f s0 = {}, s1 = {};
    s0 = wmma_bf16(qa, kb0, s0);
    s1 = wmma_bf16(qa, kb1, s1);

    int key0 = kt + nn, key1 = kt + 16 + nn;
    bool mk0 = (mask[b * SEQ + key0] == 0);
    bool mk1 = (mask[b * SEQ + key1] == 0);

#pragma unroll
    for (int r = 0; r < 8; ++r) {
      int qq = qt * 16 + mh + r;
      float b0 = Bh[(size_t)qq * SEQ + key0];   // coalesced 64B row segment
      float b1 = Bh[(size_t)qq * SEQ + key1];
      float v0 = (mq[r] || mk0) ? -1e9f : s0[r] + b0;
      float v1 = (mq[r] || mk1) ? -1e9f : s1[r] + b1;
      // row reduction across the 16 lanes of this half (row lives in one half)
      float mx = fmaxf(v0, v1);
#pragma unroll
      for (int off = 8; off; off >>= 1) mx = fmaxf(mx, __shfl_xor(mx, off, 16));
      float mnew = fmaxf(m[r], mx);
      float sc = __expf(m[r] - mnew);
      float p0 = __expf(v0 - mnew), p1 = __expf(v1 - mnew);
      float rs = p0 + p1;
#pragma unroll
      for (int off = 8; off; off >>= 1) rs += __shfl_xor(rs, off, 16);
      lsum[r] = lsum[r] * sc + rs;
      o0[r] *= sc; o1[r] *= sc;
      m[r] = mnew;
      pl[(mh + r) * 32 + nn]      = (__bf16)p0;
      pl[(mh + r) * 32 + 16 + nn] = (__bf16)p1;
    }
    __syncthreads();
    // transpose-read P into A-fragment layout: two ds b128 reads per lane
    v16bf pa;
    {
      const __bf16* pp = pl + (lane & 15) * 32 + (half << 3);
      bfpack u;
      u.q[0] = *(const u32x4*)(pp);
      u.q[1] = *(const u32x4*)(pp + 16);
      pa = u.v;
    }
    __syncthreads();
    // P (16x32 over keys) x V (32x32) -> accumulate O (16x32 over d).
    // Vt is [DH, SEQ] so B[k][d] = Vt[d][k]: contiguous per lane.
    v16bf vb0 = load_Brow(Vt, SEQ, 0, kt);
    v16bf vb1 = load_Brow(Vt, SEQ, 16, kt);
    o0 = wmma_bf16(pa, vb0, o0);
    o1 = wmma_bf16(pa, vb1, o1);
  }

#pragma unroll
  for (int r = 0; r < 8; ++r) {
    size_t row = (size_t)b * SEQ + qt * 16 + mh + r;
    float invl = 1.0f / lsum[r];
    int c0 = h * DH + nn, c1 = h * DH + 16 + nn;
    float g0 = gate[row * D_IN + c0];
    float g1 = gate[row * D_IN + c1];
    go[row * D_IN + c0] = (__bf16)(o0[r] * invl * g0);
    go[row * D_IN + c1] = (__bf16)(o1[r] * invl * g1);
  }
}

// Final projection: (gate*attn_out)[1024,256] @ Wo[256,256] + bo, mask rows.
__global__ void k_out(const __bf16* __restrict__ go, const __bf16* __restrict__ wot,
                      const float* __restrict__ bo, const int* __restrict__ mask,
                      float* __restrict__ out) {
  int mt = blockIdx.x, nt = blockIdx.y;
  v8f c = {};
#pragma unroll
  for (int k0 = 0; k0 < D_IN; k0 += 32) {
    v16bf a = load_A16x32(go, D_IN, mt * 16, k0);
    v16bf b = load_Brow(wot, D_IN, nt * 16, k0);
    c = wmma_bf16(a, b, c);
  }
  int lane = threadIdx.x & 31;
  int n  = nt * 16 + (lane & 15);
  int mh = (lane >> 4) << 3;
#pragma unroll
  for (int r = 0; r < 8; ++r) {
    int row = mt * 16 + mh + r;
    float val = (mask[row] == 0) ? 0.0f : c[r] + bo[n];
    out[(size_t)row * D_IN + n] = val;
  }
}

// ---------------------------------------------------------------- launch
extern "C" void kernel_launch(void* const* d_in, const int* in_sizes, int n_in,
                              void* d_out, int out_size, void* d_ws, size_t ws_size,
                              hipStream_t stream) {
  (void)in_sizes; (void)n_in; (void)out_size; (void)ws_size;
  const float* x       = (const float*)d_in[0];
  const float* bias    = (const float*)d_in[1];
  const int*   mask    = (const int*)  d_in[2];
  const float* ln_in_g = (const float*)d_in[3];
  const float* ln_in_b = (const float*)d_in[4];
  const float* ln_b_g  = (const float*)d_in[5];
  const float* ln_b_b  = (const float*)d_in[6];
  const float* Wq = (const float*)d_in[7];
  const float* Wk = (const float*)d_in[8];
  const float* Wv = (const float*)d_in[9];
  const float* Wb = (const float*)d_in[10];
  const float* Wg = (const float*)d_in[11];
  const float* bg = (const float*)d_in[12];
  const float* Wo = (const float*)d_in[13];
  const float* bo = (const float*)d_in[14];
  float* out = (float*)d_out;

  char* ws = (char*)d_ws;
  size_t off = 0;
  auto alloc = [&](size_t bytes) { char* p = ws + off; off += (bytes + 255) & ~(size_t)255; return p; };
  float*  xn   = (float*)  alloc((size_t)NROW * D_IN * 4);
  __bf16* xnb  = (__bf16*) alloc((size_t)NROW * D_IN * 2);
  __bf16* wqt  = (__bf16*) alloc((size_t)D_IN * D_IN * 2);
  __bf16* wkt  = (__bf16*) alloc((size_t)D_IN * D_IN * 2);
  __bf16* wvt  = (__bf16*) alloc((size_t)D_IN * D_IN * 2);
  __bf16* wgt  = (__bf16*) alloc((size_t)D_IN * D_IN * 2);
  __bf16* wot  = (__bf16*) alloc((size_t)D_IN * D_IN * 2);
  __bf16* qbuf = (__bf16*) alloc((size_t)BATCH * NH * SEQ * DH * 2);
  __bf16* kbuf = (__bf16*) alloc((size_t)BATCH * NH * SEQ * DH * 2);
  __bf16* vtb  = (__bf16*) alloc((size_t)BATCH * NH * DH * SEQ * 2);
  float*  gate = (float*)  alloc((size_t)NROW * D_IN * 4);
  __bf16* go   = (__bf16*) alloc((size_t)NROW * D_IN * 2);
  float*  bh   = (float*)  alloc((size_t)BATCH * NH * SEQ * SEQ * 4);

  // 1) LN(x) + bf16 copy
  k_ln_x<<<NROW, D_IN, 0, stream>>>(x, ln_in_g, ln_in_b, xn, xnb);
  // 2) weights -> transposed bf16
  k_cvt5t<<<(D_IN * D_IN + 255) / 256, 256, 0, stream>>>(Wq, Wk, Wv, Wg, Wo,
                                                         wqt, wkt, wvt, wgt, wot,
                                                         D_IN * D_IN);
  // 3) bias stream -> bias_h  (bandwidth-dominant; async LDS pipelined)
  k_bias<<<(BATCH * SEQ * SEQ) / (8 * BIAS_ROWS), 256, 0, stream>>>(
      bias, ln_b_g, ln_b_b, Wb, bh);
  // 4) q/k/v/gate projections (WMMA)
  {
    dim3 grid(NROW / 16, D_IN / 16, 4);
    k_qkvg<<<grid, 32, 0, stream>>>(xnb, wqt, wkt, wvt, wgt, bg,
                                    qbuf, kbuf, vtb, gate);
  }
  // 5) flash attention (WMMA)
  {
    dim3 grid(SEQ / 16, NH, BATCH);
    k_attn<<<grid, 32, 0, stream>>>(qbuf, kbuf, vtb, bh, gate, mask, go);
  }
  // 6) output projection (WMMA)
  {
    dim3 grid(NROW / 16, D_IN / 16);
    k_out<<<grid, 32, 0, stream>>>(go, wot, bo, mask, out);
  }
}
